// MetaInvH_81801947119912
// MI455X (gfx1250) — matrix operand order repaired and checked by reading, not compiled, not run on hardware
//
#include <hip/hip_runtime.h>

#define IMG 512
#define HW  262144          // 512*512
#define NPIX (8*HW)         // B=8

typedef __attribute__((ext_vector_type(16))) _Float16 v16h;
typedef __attribute__((ext_vector_type(8)))  _Float16 v8h;
typedef __attribute__((ext_vector_type(8)))  float    v8f;

// ---------------- generic small stencils (zero padded, optional spatial flip) ----------
template<int K, bool FLIP>
static __device__ __forceinline__ float convK(const float* __restrict__ img,
                                              const float* __restrict__ ker,
                                              int y, int x) {
  const int P = K / 2;
  float acc = 0.f;
#pragma unroll
  for (int dy = 0; dy < K; ++dy) {
    int yy = y + dy - P;
    if ((unsigned)yy >= (unsigned)IMG) continue;
    const float* row = img + yy * IMG;
#pragma unroll
    for (int dx = 0; dx < K; ++dx) {
      int xx = x + dx - P;
      if ((unsigned)xx >= (unsigned)IMG) continue;
      float kv = FLIP ? ker[(K - 1 - dy) * K + (K - 1 - dx)] : ker[dy * K + dx];
      acc = fmaf(row[xx], kv, acc);
    }
  }
  return acc;
}

// (A^T A + mu W^T W) applied via precomputed Ap = A p, Wp = W p
static __device__ __forceinline__ float ataw(const float* __restrict__ Apb,
                                             const float* __restrict__ Wpb,
                                             const float* __restrict__ Ka,
                                             const float* __restrict__ Kw,
                                             float mu, int y, int x) {
  float t = convK<5, true>(Apb, Ka, y, x);
  float w = 0.f;
#pragma unroll
  for (int c = 0; c < 4; ++c)
    w += convK<3, true>(Wpb + (size_t)c * HW, Kw + c * 9, y, x);
  return t + mu * w;
}

static __device__ __forceinline__ float blockSum(float v, float* sb) {
  int t = threadIdx.x;
  sb[t] = v;
  __syncthreads();
#pragma unroll
  for (int o = 128; o > 0; o >>= 1) {
    if (t < o) sb[t] += sb[t + o];
    __syncthreads();
  }
  float r = sb[0];
  __syncthreads();
  return r;
}

// ---------------- WMMA fragment loaders (LDS) ----------------
static __device__ __forceinline__ v16h cat8(v8h lo, v8h hi) {
  return __builtin_shufflevector(lo, hi, 0, 1, 2, 3, 4, 5, 6, 7,
                                 8, 9, 10, 11, 12, 13, 14, 15);
}
// A 16x32 f16: lane<16 -> K{0..7,16..23}, lane>=16 -> K{8..15,24..31} (caller offsets base)
static __device__ __forceinline__ v16h ldA(const _Float16* p) {
  v8h lo = *(const v8h*)p;
  v8h hi = *(const v8h*)(p + 16);
  return cat8(lo, hi);
}
// B 32x16 f16: lane<16 -> K{0..7,16..23} of its pixel, lane>=16 -> K{8..15,24..31} (caller offsets base)
static __device__ __forceinline__ v16h ldB(const _Float16* p) {
  v8h lo = *(const v8h*)p;
  v8h hi = *(const v8h*)(p + 16);
  return cat8(lo, hi);
}

// ---------------- fused CNN: u + conv3(relu(conv2(relu(conv1(u))))) --------------------
// tile: 16 wide x 8 tall; u tile 14x36 (halo 3); h1 12x34x32 f16; h2 10x18x32 f16
__global__ __launch_bounds__(256) void k_cnn(const float* __restrict__ xin,
                                             const float* __restrict__ W1,
                                             const float* __restrict__ b1,
                                             const float* __restrict__ W2,
                                             const float* __restrict__ b2,
                                             const float* __restrict__ W3,
                                             const float* __restrict__ b3,
                                             float* __restrict__ out) {
  __shared__ __align__(16) float us[14 * 36];
  __shared__ __align__(16) _Float16 h1s[12 * 34 * 32];
  __shared__ __align__(16) _Float16 h2s[10 * 18 * 32];
  __shared__ __align__(16) _Float16 w2s[9 * 32 * 32];   // [tap][co][ci]
  __shared__ float w1s[288], w3s[288], b1s[32], b2s[32];
  __shared__ float b3s;

  const int tid = threadIdx.x;
  const int bi  = blockIdx.x;
  const int b   = bi >> 11;            // 2048 tiles per batch image
  const int rem = bi & 2047;
  const int ty  = rem >> 5;            // 64 tile-rows
  const int tx  = rem & 31;            // 32 tile-cols
  const int gy0 = ty * 8, gx0 = tx * 16;
  const float* xb = xin + (size_t)b * HW;

  // ---- stage input tile ----
  // Interior tiles (halo fully in-bounds): CDNA5 async global->LDS copy, per-wave
  // ASYNCcnt wait. Boundary tiles: manual zero-padded path.
  const bool interior = (ty >= 1) && (ty <= 62) && (tx >= 1) && (tx <= 29);
  if (interior) {
    for (int i = tid; i < 14 * 36; i += 256) {
      int rr = i / 36, cc = i % 36;
      unsigned lds = (unsigned)(uintptr_t)(&us[i]);                 // LDS byte offset
      unsigned voff = (unsigned)(((gy0 + rr - 3) * IMG + (gx0 + cc - 3)) * 4);
      asm volatile("global_load_async_to_lds_b32 %0, %1, %2"
                   :: "v"(lds), "v"(voff), "s"(xb) : "memory");
    }
    asm volatile("s_wait_asynccnt 0x0" ::: "memory");
  } else {
    for (int i = tid; i < 14 * 36; i += 256) {
      int rr = i / 36, cc = i % 36;
      int gy = gy0 + rr - 3, gx = gx0 + cc - 3;
      float v = 0.f;
      if ((unsigned)gy < (unsigned)IMG && (unsigned)gx < (unsigned)IMG) v = xb[gy * IMG + gx];
      us[i] = v;
    }
  }
  // stage weights; W2 reorganized to [tap][co][ci] in f16
  for (int i = tid; i < 288; i += 256) { w1s[i] = W1[i]; w3s[i] = W3[i]; }
  if (tid < 32) { b1s[tid] = b1[tid]; b2s[tid] = b2[tid]; }
  if (tid == 0) b3s = b3[0];
  for (int l = tid; l < 9216; l += 256) {
    int t = l >> 10, co = (l >> 5) & 31, ci = l & 31;
    w2s[l] = (_Float16)W2[co * 288 + ci * 9 + t];
  }
  __syncthreads();

  // conv1 (1->32) + ReLU -> h1 (f16, [pixel][ci]) with b128 stores
  for (int pidx = tid; pidx < 12 * 34; pidx += 256) {
    int row = pidx / 34, col = pidx % 34;
    float u[9];
#pragma unroll
    for (int t = 0; t < 9; ++t)
      u[t] = us[(row + t / 3) * 36 + (col + t % 3)];
    _Float16* hp = &h1s[pidx * 32];
#pragma unroll
    for (int cb = 0; cb < 4; ++cb) {
      v8h hv;
#pragma unroll
      for (int k = 0; k < 8; ++k) {
        int c = cb * 8 + k;
        float a = b1s[c];
#pragma unroll
        for (int t = 0; t < 9; ++t) a = fmaf(u[t], w1s[c * 9 + t], a);
        hv[k] = (_Float16)fmaxf(a, 0.f);
      }
      *(v8h*)(hp + cb * 8) = hv;
    }
  }
  __syncthreads();

  // conv2 (32->32) via WMMA: 10 h2-rows x 2 groups of 16 pixels, 9 taps, 2 M-halves
  const int wid = tid >> 5, lane = tid & 31;
  const int lsel = (lane < 16) ? 0 : 1;
  const int lm = lane & 15;
  for (int job = wid; job < 20; job += 8) {
    const int j = job >> 1, g = job & 1;   // h2 storage row j (actual row j-1), pixel group g
    v8f c0 = {};
    v8f c1 = {};
#pragma unroll
    for (int t = 0; t < 9; ++t) {
      const int dy = t / 3, dx = t % 3;
      const _Float16* bp = &h1s[(((j + dy) * 34) + (g * 16 + dx + lm)) * 32 + lsel * 8];
      v16h bf = ldB(bp);
      const _Float16* a0 = &w2s[(t * 32 + lm) * 32 + lsel * 8];
      c0 = __builtin_amdgcn_wmma_f32_16x16x32_f16(false, ldA(a0), false, bf,
                                                  (short)0, c0, false, false);
      const _Float16* a1 = &w2s[(t * 32 + 16 + lm) * 32 + lsel * 8];
      c1 = __builtin_amdgcn_wmma_f32_16x16x32_f16(false, ldA(a1), false, bf,
                                                  (short)0, c1, false, false);
    }
    // bias + ReLU, store h2 (clip overhanging pixel columns)
    const int cst = g * 16 + lm;
    if (cst < 18) {
      _Float16* hp = &h2s[(j * 18 + cst) * 32];
#pragma unroll
      for (int v = 0; v < 8; ++v) {
        int ch0 = v + lsel * 8;            // C/D layout: VGPR v, lanes>=16 are M+8
        hp[ch0]      = (_Float16)fmaxf(c0[v] + b2s[ch0], 0.f);
        hp[16 + ch0] = (_Float16)fmaxf(c1[v] + b2s[16 + ch0], 0.f);
      }
    }
  }
  __syncthreads();

  // conv3 (32->1) + residual, vectorized b128 LDS reads
  if (tid < 128) {
    int py = tid >> 4, px = tid & 15;
    float acc = b3s;
#pragma unroll
    for (int t = 0; t < 9; ++t) {
      const _Float16* hp = &h2s[((py + t / 3) * 18 + (px + t % 3)) * 32];
#pragma unroll
      for (int cb = 0; cb < 4; ++cb) {
        v8h hv = *(const v8h*)(hp + cb * 8);
#pragma unroll
        for (int k = 0; k < 8; ++k)
          acc = fmaf((float)hv[k], w3s[(cb * 8 + k) * 9 + t], acc);
      }
    }
    int gy = gy0 + py, gx = gx0 + px;
    out[(size_t)b * HW + gy * IMG + gx] = us[(py + 3) * 36 + (px + 3)] + acc;
  }
}

// ---------------- CG-phase kernels -------------------------------------------------------
// dnz = softthresh(conv(x,Kw), laam) -> stored in Wp buffer
__global__ void k_dnz(const float* __restrict__ x, const float* __restrict__ Kw,
                      const float* __restrict__ laam, float* __restrict__ dnz) {
  int b = blockIdx.y;
  int idx = blockIdx.x * 256 + threadIdx.x;   // 0 .. 4*HW-1
  int c = idx >> 18, pix = idx & (HW - 1);
  int y = pix >> 9, xx = pix & 511;
  float wu = convK<3, false>(x + (size_t)b * HW, Kw + c * 9, y, xx);
  float lam = laam[0];
  float d = fmaxf(wu - lam, 0.f) - fmaxf(-wu - lam, 0.f);
  dnz[((size_t)b * 4 + c) * HW + pix] = d;
}

// rhs = ATop(sino) + mu * WTop(dnz)
__global__ void k_rhs(const float* __restrict__ sino, const float* __restrict__ dnz,
                      const float* __restrict__ Ka, const float* __restrict__ Kw,
                      const float* __restrict__ miu, float* __restrict__ rhs) {
  int b = blockIdx.y;
  int pix = blockIdx.x * 256 + threadIdx.x;
  int y = pix >> 9, x = pix & 511;
  float s = convK<5, true>(sino + (size_t)b * HW, Ka, y, x);
  float wt = 0.f;
#pragma unroll
  for (int c = 0; c < 4; ++c)
    wt += convK<3, true>(dnz + ((size_t)b * 4 + c) * HW, Kw + c * 9, y, x);
  rhs[(size_t)b * HW + pix] = s + miu[0] * wt;
}

// Ap = A p, Wp = W p ; partials of {Ap.Ap, mu*Wp.Wp, r.r}
__global__ void k_ApWp(const float* __restrict__ pin, const float* __restrict__ r,
                       const float* __restrict__ Ka, const float* __restrict__ Kw,
                       const float* __restrict__ miu,
                       float* __restrict__ Ap, float* __restrict__ Wp,
                       float* __restrict__ parts) {
  __shared__ float sb[256];
  int b = blockIdx.y;
  int pix = blockIdx.x * 256 + threadIdx.x;
  int y = pix >> 9, x = pix & 511;
  const float* pb = pin + (size_t)b * HW;
  float ap = convK<5, false>(pb, Ka, y, x);
  Ap[(size_t)b * HW + pix] = ap;
  float wsq = 0.f;
#pragma unroll
  for (int c = 0; c < 4; ++c) {
    float w = convK<3, false>(pb, Kw + c * 9, y, x);
    Wp[((size_t)b * 4 + c) * HW + pix] = w;
    wsq = fmaf(w, w, wsq);
  }
  float rv = r[(size_t)b * HW + pix];
  float s0 = blockSum(ap * ap, sb);
  float s1 = blockSum(miu[0] * wsq, sb);
  float s2 = blockSum(rv * rv, sb);
  if (threadIdx.x == 0) {
    float* q = parts + ((size_t)b * 1024 + blockIdx.x) * 3;
    q[0] = s0; q[1] = s1; q[2] = s2;
  }
}

__global__ void k_reduce_al(const float* __restrict__ parts, float* __restrict__ scal) {
  __shared__ float sb[256];
  int b = blockIdx.x, t = threadIdx.x;
  float a = 0.f, w = 0.f, rr = 0.f;
  for (int i = t; i < 1024; i += 256) {
    const float* q = parts + ((size_t)b * 1024 + i) * 3;
    a += q[0]; w += q[1]; rr += q[2];
  }
  a = blockSum(a, sb); w = blockSum(w, sb); rr = blockSum(rr, sb);
  if (t == 0) { scal[b * 4 + 0] = rr / (a + w); scal[b * 4 + 1] = rr; }
}

// r,p initialization: res = (A^T A + mu W^T W) uk0 - rhs
__global__ void k_init_r(const float* __restrict__ Ap, const float* __restrict__ Wp,
                         const float* __restrict__ Ka, const float* __restrict__ Kw,
                         const float* __restrict__ miu, const float* __restrict__ rhs,
                         float* __restrict__ r, float* __restrict__ p) {
  int b = blockIdx.y;
  int pix = blockIdx.x * 256 + threadIdx.x;
  int y = pix >> 9, x = pix & 511;
  float t = ataw(Ap + (size_t)b * HW, Wp + (size_t)b * 4 * HW, Ka, Kw, miu[0], y, x);
  size_t i = (size_t)b * HW + pix;
  float res = t - rhs[i];
  r[i] = res;
  p[i] = -res;
}

// xk += al*p ; r += al*(ATop(Ap)+mu*WTop(Wp)) ; partials of new r.r
__global__ void k_update(const float* __restrict__ Ap, const float* __restrict__ Wp,
                         const float* __restrict__ Ka, const float* __restrict__ Kw,
                         const float* __restrict__ miu, const float* __restrict__ scal,
                         float* __restrict__ r, float* __restrict__ xk,
                         const float* __restrict__ p, float* __restrict__ parts) {
  __shared__ float sb[256];
  int b = blockIdx.y;
  int pix = blockIdx.x * 256 + threadIdx.x;
  int y = pix >> 9, x = pix & 511;
  float al = scal[b * 4 + 0];
  float t = ataw(Ap + (size_t)b * HW, Wp + (size_t)b * 4 * HW, Ka, Kw, miu[0], y, x);
  size_t i = (size_t)b * HW + pix;
  float rn = fmaf(al, t, r[i]);
  xk[i] = fmaf(al, p[i], xk[i]);
  r[i] = rn;
  float s = blockSum(rn * rn, sb);
  if (threadIdx.x == 0) parts[((size_t)b * 1024 + blockIdx.x) * 3] = s;
}

__global__ void k_reduce_be(const float* __restrict__ parts, float* __restrict__ scal) {
  __shared__ float sb[256];
  int b = blockIdx.x, t = threadIdx.x;
  float rn = 0.f;
  for (int i = t; i < 1024; i += 256) rn += parts[((size_t)b * 1024 + i) * 3];
  rn = blockSum(rn, sb);
  if (t == 0) { scal[b * 4 + 2] = rn / scal[b * 4 + 1]; scal[b * 4 + 1] = rn; }
}

__global__ void k_pupdate(const float* __restrict__ r, const float* __restrict__ scal,
                          float* __restrict__ p) {
  int b = blockIdx.y;
  size_t i = (size_t)b * HW + blockIdx.x * 256 + threadIdx.x;
  p[i] = fmaf(scal[b * 4 + 2], p[i], -r[i]);
}

__global__ void k_final(const float* __restrict__ xk, const float* __restrict__ p,
                        const float* __restrict__ scal, float* __restrict__ out) {
  int b = blockIdx.y;
  size_t i = (size_t)b * HW + blockIdx.x * 256 + threadIdx.x;
  out[i] = fmaf(scal[b * 4 + 0], p[i], xk[i]);
}

// ---------------- driver ---------------------------------------------------------------
extern "C" void kernel_launch(void* const* d_in, const int* in_sizes, int n_in,
                              void* d_out, int out_size, void* d_ws, size_t ws_size,
                              hipStream_t stream) {
  (void)in_sizes; (void)n_in; (void)out_size; (void)ws_size;
  const float* sino = (const float*)d_in[0];
  const float* x    = (const float*)d_in[1];
  const float* laam = (const float*)d_in[2];
  const float* miu  = (const float*)d_in[3];
  const float* Kw   = (const float*)d_in[4];
  const float* Ka   = (const float*)d_in[5];
  const float* W1   = (const float*)d_in[6];
  const float* b1   = (const float*)d_in[7];
  const float* W2   = (const float*)d_in[8];
  const float* b2   = (const float*)d_in[9];
  const float* W3   = (const float*)d_in[10];
  const float* b3   = (const float*)d_in[11];

  float* rhs   = (float*)d_ws;
  float* xk    = rhs + NPIX;
  float* r     = xk + NPIX;
  float* p     = r + NPIX;
  float* Ap    = p + NPIX;
  float* Wp    = Ap + NPIX;            // 4*NPIX; also holds dnz before CG
  float* parts = Wp + (size_t)4 * NPIX;
  float* scal  = parts + (size_t)8 * 1024 * 3;

  dim3 blk(256);
  dim3 gpix(1024, 8);

  k_dnz<<<dim3(4096, 8), blk, 0, stream>>>(x, Kw, laam, Wp);
  k_rhs<<<gpix, blk, 0, stream>>>(sino, Wp, Ka, Kw, miu, rhs);
  k_cnn<<<16384, blk, 0, stream>>>(x, W1, b1, W2, b2, W3, b3, xk);   // xk = uk0
  k_ApWp<<<gpix, blk, 0, stream>>>(xk, xk, Ka, Kw, miu, Ap, Wp, parts);  // A uk0, W uk0
  k_init_r<<<gpix, blk, 0, stream>>>(Ap, Wp, Ka, Kw, miu, rhs, r, p);

  for (int it = 0; it < 5; ++it) {
    k_ApWp<<<gpix, blk, 0, stream>>>(p, r, Ka, Kw, miu, Ap, Wp, parts);
    k_reduce_al<<<8, blk, 0, stream>>>(parts, scal);
    k_update<<<gpix, blk, 0, stream>>>(Ap, Wp, Ka, Kw, miu, scal, r, xk, p, parts);
    k_reduce_be<<<8, blk, 0, stream>>>(parts, scal);
    k_pupdate<<<gpix, blk, 0, stream>>>(r, scal, p);
  }
  // final half step
  k_ApWp<<<gpix, blk, 0, stream>>>(p, r, Ka, Kw, miu, Ap, Wp, parts);
  k_reduce_al<<<8, blk, 0, stream>>>(parts, scal);
  k_final<<<gpix, blk, 0, stream>>>(xk, p, scal, (float*)d_out);
}
